// EdgeAtt_15092515078264
// MI455X (gfx1250) — matrix operand order repaired and checked by reading, compiled, not run on hardware
//
#include <hip/hip_runtime.h>

#define B_  32
#define L_  1024
#define G_  512
#define WP_ 6
#define WW_ 13
#define SCALE_ 0.07071067811865475f   // 1/sqrt(200)

typedef __attribute__((ext_vector_type(16))) __bf16 v16bf;
typedef __attribute__((ext_vector_type(8)))  float  v8f;
typedef __attribute__((ext_vector_type(4)))  float  v4f;

// ---------------------------------------------------------------------------
// Stage 1: A[m, g] = sum_d W[g, d] * nf[m, d]   (m = b*L + l, M = 32768)
// bf16 WMMA 16x16x32, fp32 accumulate, fp32 -> bf16 conversion on the fly.
// Block = 256 threads = 8 waves; wave w owns n-tile (blockIdx.y*128 + 16w);
// all 8 waves share one 16-row m-tile (A-operand hits L0/L2).
// K-loop unrolled x4 so 4 WMMAs + their loads are in flight per iteration.
// ---------------------------------------------------------------------------
__global__ __launch_bounds__(256) void gemm_wmma_kernel(
    const float* __restrict__ nf,   // [B*L, G]
    const float* __restrict__ w,    // [G, G]
    float* __restrict__ A)          // [B*L, G]  (workspace)
{
    const int lane   = threadIdx.x & 31;
    const int wave   = threadIdx.x >> 5;              // 0..7
    const int m_base = blockIdx.x * 16;               // 2048 m-tiles
    const int n_base = blockIdx.y * 128 + wave * 16;  // 4 * 8 n-tiles

    const int half = lane >> 4;    // 0: lanes 0-15, 1: lanes 16-31
    const int l15  = lane & 15;

    const float* arow = nf + (size_t)(m_base + l15) * G_;
    const float* brow = w  + (size_t)(n_base + l15) * G_;

    v8f acc = {};
#pragma unroll 4
    for (int k0 = 0; k0 < G_; k0 += 32) {
        // A-operand (16x32): row m_base+l15, K chunks {ad..ad+7, ad+16..ad+23}
        const int ad = k0 + half * 8;
        v4f a0 = *(const v4f*)(arow + ad);
        v4f a1 = *(const v4f*)(arow + ad + 4);
        v4f a2 = *(const v4f*)(arow + ad + 16);
        v4f a3 = *(const v4f*)(arow + ad + 20);
        // B-operand (32x16): B[k][n] = W[n][k]; K chunks {bd..bd+7, bd+16..bd+23} from W row n
        const int bd = k0 + half * 8;
        v4f b0 = *(const v4f*)(brow + bd);
        v4f b1 = *(const v4f*)(brow + bd + 4);
        v4f b2 = *(const v4f*)(brow + bd + 16);
        v4f b3 = *(const v4f*)(brow + bd + 20);

        v16bf av, bv;
#pragma unroll
        for (int i = 0; i < 4; ++i) {
            av[i]      = (__bf16)a0[i];
            av[4 + i]  = (__bf16)a1[i];
            av[8 + i]  = (__bf16)a2[i];
            av[12 + i] = (__bf16)a3[i];
            bv[i]      = (__bf16)b0[i];
            bv[4 + i]  = (__bf16)b1[i];
            bv[8 + i]  = (__bf16)b2[i];
            bv[12 + i] = (__bf16)b3[i];
        }
        // (neg_a, A, neg_b, B, c_mod, C, reuse_a, reuse_b)
        acc = __builtin_amdgcn_wmma_f32_16x16x32_bf16(
            false, av, false, bv, (short)0, acc, false, false);
    }

    // D layout: VGPR v -> row (v + half*8), col l15 -> coalesced fp32 stores
    float* crow = A + (size_t)(m_base + half * 8) * G_ + n_base + l15;
#pragma unroll
    for (int v = 0; v < 8; ++v)
        crow[(size_t)v * G_] = acc[v];
}

// ---------------------------------------------------------------------------
// Stage 2 (fused): one wave per (b,l):
//   s[w] = SCALE * nf[b,l,:] . A[b, clip(l-6+w), :]  (lanes split G, float4)
//   masked softmax over 13 offsets, then the wave zero-fills its whole
//   1024-float alpha row (b128 stores) and lane 0 overwrites the 13-wide
//   band. Same-wave stores are ordered, so no extra sync is needed.
// ---------------------------------------------------------------------------
__global__ __launch_bounds__(256) void score_softmax_kernel(
    const float* __restrict__ nf,        // [B, L, G]
    const int*   __restrict__ node_num,  // [B]
    const float* __restrict__ A,         // [B*L, G]
    float* __restrict__ alpha)           // [B, L, L]
{
    const int lane = threadIdx.x & 31;
    const int gw   = (int)((blockIdx.x * blockDim.x + threadIdx.x) >> 5);
    if (gw >= B_ * L_) return;
    const int b   = gw >> 10;        // / L
    const int l   = gw & (L_ - 1);
    const int cur = node_num[b];

    const float* xrow = nf + ((size_t)b * L_ + l) * G_;
    v4f xr[4];
#pragma unroll
    for (int i = 0; i < 4; ++i)
        xr[i] = *(const v4f*)(xrow + lane * 4 + i * 128);

    float sc[WW_];
#pragma unroll
    for (int w = 0; w < WW_; ++w) {
        const int raw = l - WP_ + w;
        const int idx = raw < 0 ? 0 : (raw > L_ - 1 ? L_ - 1 : raw);
        const float* ar = A + ((size_t)b * L_ + idx) * G_;
        float s = 0.f;
#pragma unroll
        for (int i = 0; i < 4; ++i) {
            v4f a = *(const v4f*)(ar + lane * 4 + i * 128);
            s += a[0] * xr[i][0] + a[1] * xr[i][1]
               + a[2] * xr[i][2] + a[3] * xr[i][3];
        }
#pragma unroll
        for (int off = 16; off > 0; off >>= 1)
            s += __shfl_xor(s, off, 32);
        sc[w] = s;
    }

    // Zero-fill this row of alpha cooperatively: 32 lanes x 8 x float4 = 1024
    float* out = alpha + ((size_t)b * L_ + l) * L_;
    {
        const v4f z = {};
#pragma unroll
        for (int i = 0; i < 8; ++i)
            *(v4f*)(out + lane * 4 + i * 128) = z;
    }

    if (lane == 0) {
        const bool rowv = (l <= cur - 1);
        bool valid[WW_];
        float m = -3.4e38f;
#pragma unroll
        for (int w = 0; w < WW_; ++w) {
            const int raw = l - WP_ + w;
            valid[w] = rowv && (raw >= 0) && (raw <= cur - 1);
            sc[w] = valid[w] ? sc[w] * SCALE_ : -1.0e9f;
            m = fmaxf(m, sc[w]);
        }
        float sum = 0.f;
#pragma unroll
        for (int w = 0; w < WW_; ++w) { sc[w] = __expf(sc[w] - m); sum += sc[w]; }
        const float inv = 1.f / sum;
#pragma unroll
        for (int w = 0; w < WW_; ++w) {
            const int raw = l - WP_ + w;
            if (valid[w]) out[raw] = sc[w] * inv;
        }
    }
}

// ---------------------------------------------------------------------------
extern "C" void kernel_launch(void* const* d_in, const int* in_sizes, int n_in,
                              void* d_out, int out_size, void* d_ws, size_t ws_size,
                              hipStream_t stream)
{
    (void)in_sizes; (void)n_in; (void)out_size; (void)ws_size;
    const float* nf = (const float*)d_in[0];   // node_features [B,L,G]
    const int*   nn = (const int*)d_in[1];     // node_num [B]
    const float* w  = (const float*)d_in[2];   // weight [G,G]
    float* alpha = (float*)d_out;              // [B,L,L]
    float* A     = (float*)d_ws;               // [B*L, G] fp32 (64 MB)

    // 1) GEMM via bf16 WMMA: 2048 m-tiles x 4 n-blocks, 8 waves/block
    dim3 gg(B_ * L_ / 16, G_ / 128);
    gemm_wmma_kernel<<<gg, 256, 0, stream>>>(nf, w, A);

    // 2) scores + softmax + fused zero-fill/band write: one wave per (b,l)
    const int waves = B_ * L_;
    score_softmax_kernel<<<waves * 32 / 256, 256, 0, stream>>>(nf, nn, A, alpha);
}
